// SecondOrderDeformableAlignment_4269197492313
// MI455X (gfx1250) — matrix operand order repaired and checked by reading, not compiled, not run on hardware
//
#include <hip/hip_runtime.h>
#include <math.h>

// ---------------- types ----------------
typedef __attribute__((ext_vector_type(16))) __bf16          v16bf;
typedef __attribute__((ext_vector_type(16))) unsigned short  v16u;
typedef __attribute__((ext_vector_type(8)))  float           v8f;
typedef __attribute__((ext_vector_type(8)))  unsigned short  v8u;
typedef __attribute__((ext_vector_type(4)))  float           v4f;
typedef int v2i __attribute__((__vector_size__(2 * sizeof(int))));  // matches builtin proto

#define B_     2
#define H_     128
#define W_     128
#define C1RAW  196     // 3*64 + 2 + 2
#define C1PAD  224     // padded to multiple of 32
#define COUT4  432     // 27*DG
#define MT4    27
#define MT4PAD 28

// ---- CDNA5 async global->LDS path (guarded; falls back to load+ds_store) ----
#if defined(__has_builtin)
# if __has_builtin(__builtin_amdgcn_global_load_async_to_lds_b64)
#  define HAVE_ASYNC_LDS 1
# endif
#endif
#ifndef HAVE_ASYNC_LDS
# define HAVE_ASYNC_LDS 0
#endif

#define GAS __attribute__((address_space(1)))
#define LAS __attribute__((address_space(3)))

static __device__ __forceinline__ void wait_async_all() {
#if defined(__has_builtin) && __has_builtin(__builtin_amdgcn_s_wait_asynccnt)
  __builtin_amdgcn_s_wait_asynccnt(0);
#else
  asm volatile("s_wait_asynccnt 0x0" ::: "memory");
#endif
}

static __device__ __forceinline__ unsigned short f2bf(float f) {
  unsigned u = __builtin_bit_cast(unsigned, f);
  unsigned r = u + 0x7FFFu + ((u >> 16) & 1u);   // round to nearest even
  return (unsigned short)(r >> 16);
}

// ------------- pack concat(extra_feat, flow1, flow2) -> bf16 NHWC[224] -------------
__global__ __launch_bounds__(256) void pack_concat_kernel(
    const float* __restrict__ ef, const float* __restrict__ f1,
    const float* __restrict__ f2, unsigned short* __restrict__ out)
{
  int idx = blockIdx.x * 256 + threadIdx.x;
  const int total = B_ * H_ * W_ * C1PAD;
  if (idx >= total) return;
  int ch  = idx % C1PAD;
  int pix = idx / C1PAD;
  int w = pix % W_, h = (pix / W_) % H_, b = pix / (W_ * H_);
  float v = 0.f;
  if (ch < 192)      v = ef[((b * 192 + ch)        * H_ + h) * W_ + w];
  else if (ch < 194) v = f1[((b * 2 + (ch - 192))  * H_ + h) * W_ + w];
  else if (ch < 196) v = f2[((b * 2 + (ch - 194))  * H_ + h) * W_ + w];
  out[idx] = f2bf(v);
}

// ------------- repack conv weight (Cout,Cin,3,3) f32 -> A-fragment-ordered bf16 -------------
// dst layout: [mtile][cblk][tap][lane(32)][e(16)]  (contiguous k = 16*(lane/16) + e, same order as the B fragment)
__global__ __launch_bounds__(256) void repack_w_kernel(
    const float* __restrict__ w, unsigned short* __restrict__ dst,
    int CoutReal, int CinReal, int cblks, int mtPad)
{
  int idx = blockIdx.x * 256 + threadIdx.x;
  int total = mtPad * cblks * 9 * 512;
  if (idx >= total) return;
  int e    = idx & 15;
  int lane = (idx >> 4) & 31;
  int t    = idx >> 9;
  int tap  = t % 9;  t /= 9;
  int cb   = t % cblks;
  int mt   = t / cblks;
  int hi = lane >> 4, m = lane & 15;
  int j = e >> 1, hf = e & 1;
  int kk = 2 * j + hf + (hi ? 16 : 0);
  int cin  = cb * 32 + kk;
  int cout = mt * 16 + m;
  float v = 0.f;
  if (cout < CoutReal && cin < CinReal)
    v = w[((cout * CinReal + cin) * 3 + tap / 3) * 3 + tap % 3];
  dst[idx] = f2bf(v);
}

// ------------- implicit-GEMM 3x3 conv via WMMA bf16 (f32 accum) -------------
// block: 256 thr = 8 waves; tile 64(M) x 32(N pixels along W)
// double-buffered LDS slab [2][3x34x32ch], filled via async global->LDS copies
// EPI 0: out = lrelu(conv+b) -> bf16 NHWC[64]
// EPI 1: conv4 epilogue -> 10*tanh(+b)+flow into offs NHWC[288], sigmoid into mask NHWC[144]
template <int EPI>
__global__ __launch_bounds__(256) void conv_wmma_kernel(
    const unsigned short* __restrict__ actIn, const unsigned short* __restrict__ wpack,
    const float* __restrict__ bias, int cblks, int mtReal,
    unsigned short* __restrict__ actOut,
    float* __restrict__ offs, float* __restrict__ maskb,
    const float* __restrict__ f1, const float* __restrict__ f2)
{
  __shared__ unsigned short slab[2][3 * 34 * 32];   // 2 x 3264 bf16 = 12.75 KB
  const int CinPad = cblks * 32;

  int nb   = blockIdx.x;                            // B*H*(W/32) = 1024
  int w0   = (nb & 3) * 32;
  int h    = (nb >> 2) & 127;
  int b    = nb >> 9;
  int tid  = threadIdx.x;
  int lane = tid & 31;
  int wave = tid >> 5;
  int mt   = blockIdx.y * 4 + (wave >> 1);
  int nt   = wave & 1;

  v8f acc = {0.f, 0.f, 0.f, 0.f, 0.f, 0.f, 0.f, 0.f};

  // issue the (async) fill of one slab buffer for channel-block cb
  auto issue_fill = [&](int cb, int buf) {
    for (int e = tid; e < 816; e += 256) {          // 816 x 8-byte chunks
      int ch4 = (e & 7) * 4;
      int col = (e >> 3) % 34;
      int row = (e >> 3) / 34;
      int hs = h + row - 1, wsx = w0 + col - 1;
      unsigned short* dst = &slab[buf][(row * 34 + col) * 32 + ch4];
      if (hs >= 0 && hs < H_ && wsx >= 0 && wsx < W_) {
        const unsigned short* p =
            actIn + (size_t)((b * H_ + hs) * W_ + wsx) * CinPad + cb * 32 + ch4;
#if HAVE_ASYNC_LDS
        __builtin_amdgcn_global_load_async_to_lds_b64(
            (GAS v2i*)p, (LAS v2i*)dst, 0, 0);
#else
        *(uint2*)dst = *(const uint2*)p;
        if (cb + 1 < cblks) __builtin_prefetch(p + 32, 0, 1);
#endif
      } else {
        uint2 z; z.x = 0u; z.y = 0u;                // zero halo via DS store
        *(uint2*)dst = z;
      }
    }
  };

  issue_fill(0, 0);
  for (int cb = 0; cb < cblks; ++cb) {
    int buf = cb & 1;
#if HAVE_ASYNC_LDS
    wait_async_all();          // own async fills of slab[buf] landed
#endif
    __syncthreads();           // everyone's fills/zeroes of slab[buf] visible
    if (cb + 1 < cblks)
      issue_fill(cb + 1, buf ^ 1);   // overlap next fill with WMMA below

    const unsigned short* wp = wpack + (size_t)(mt * cblks + cb) * 9 * 512;
#pragma unroll
    for (int tap = 0; tap < 9; ++tap) {
      int ky = tap / 3, kx = tap % 3;
      // A fragment: fragment-ordered weights, 32B contiguous per lane
      v16bf a = __builtin_bit_cast(v16bf, *(const v16u*)(wp + tap * 512 + lane * 16));
      // B fragment: lanes 0-15 -> K=ch 0..15, lanes 16-31 -> K=ch 16..31 (32B per lane)
      int colIdx = (lane & 15) + nt * 16 + kx;
      v16bf bf = __builtin_bit_cast(
          v16bf, *(const v16u*)&slab[buf][(ky * 34 + colIdx) * 32 + (lane >> 4) * 16]);
      acc = __builtin_amdgcn_wmma_f32_16x16x32_bf16(false, a, false, bf,
                                                    (short)0, acc, false, false);
    }
  }

  if (mt >= mtReal) return;
  int n = lane & 15, hi = lane >> 4;
  int p = w0 + nt * 16 + n;
  size_t pixbase = (size_t)((b * H_ + h) * W_ + p);
  int chbase = mt * 16 + hi * 8;                 // C/D layout: VGPR r -> M = r + 8*hi

  if (EPI == 0) {
    v8u pk;
#pragma unroll
    for (int r = 0; r < 8; ++r) {
      float v = acc[r] + bias[chbase + r];
      v = v >= 0.f ? v : 0.1f * v;
      pk[r] = f2bf(v);
    }
    *(v8u*)&actOut[pixbase * 64 + chbase] = pk;  // 16B contiguous
  } else {
    if (chbase >= 288) {                         // mask region (sigmoid)
      v4f lo, hi4;
#pragma unroll
      for (int r = 0; r < 8; ++r) {
        float v = acc[r] + bias[chbase + r];
        float s = 1.f / (1.f + __expf(-v));
        if (r < 4) lo[r] = s; else hi4[r - 4] = s;
      }
      float* dp = &maskb[pixbase * 144 + (chbase - 288)];
      *(v4f*)dp = lo; *(v4f*)(dp + 4) = hi4;
    } else {                                     // offset region (10*tanh + flow)
      const float* fl = (chbase < 144) ? f1 : f2;
      v4f lo, hi4;
#pragma unroll
      for (int r = 0; r < 8; ++r) {
        int comp = (chbase + r) & 1;
        float flow = fl[((b * 2 + comp) * H_ + h) * W_ + p];
        float v = 10.f * tanhf(acc[r] + bias[chbase + r]) + flow;
        if (r < 4) lo[r] = v; else hi4[r - 4] = v;
      }
      float* dp = &offs[pixbase * 288 + chbase];
      *(v4f*)dp = lo; *(v4f*)(dp + 4) = hi4;
    }
  }
}

// ------------- modulated deformable conv (gather + VALU GEMM) -------------
// block handles 16 pixels of one row; LDS holds val[16][1152] (cin*9+tap order)
__global__ __launch_bounds__(256) void dcn_kernel(
    const float* __restrict__ x, const float* __restrict__ offs,
    const float* __restrict__ maskb, const float* __restrict__ wdcn,
    const float* __restrict__ bdcn, float* __restrict__ out)
{
  __shared__ float val[16 * 1152];                // 73.7 KB (320 KB LDS per WGP)
  int pb = blockIdx.x;                            // B*H*(W/16) = 2048
  int w0 = (pb & 7) * 16;
  int h  = (pb >> 3) & 127;
  int b  = pb >> 10;
  int tid = threadIdx.x;

  // phase 1: bilinear sampling into LDS
  for (int j = tid; j < 2304; j += 256) {         // 16 px * 16 g * 9 k
    int k = j % 9;
    int g = (j / 9) & 15;
    int p = j / 144;
    int w = w0 + p;
    size_t pix = (size_t)((b * H_ + h) * W_ + w);
    float oy = offs[pix * 288 + g * 18 + k * 2 + 0];
    float ox = offs[pix * 288 + g * 18 + k * 2 + 1];
    float mv = maskb[pix * 144 + g * 9 + k];
    float sy = (float)(h + k / 3 - 1) + oy;
    float sx = (float)(w + k % 3 - 1) + ox;
    float y0f = floorf(sy), x0f = floorf(sx);
    int y0 = (int)y0f, x0 = (int)x0f;
    float fy = sy - y0f, fx = sx - x0f;
    float w00 = (1.f - fy) * (1.f - fx), w01 = (1.f - fy) * fx;
    float w10 = fy * (1.f - fx),         w11 = fy * fx;
    bool vy0 = (y0 >= 0) & (y0 < H_),     vy1 = (y0 + 1 >= 0) & (y0 + 1 < H_);
    bool vx0 = (x0 >= 0) & (x0 < W_),     vx1 = (x0 + 1 >= 0) & (x0 + 1 < W_);
    w00 *= (vy0 && vx0) ? 1.f : 0.f;  w01 *= (vy0 && vx1) ? 1.f : 0.f;
    w10 *= (vy1 && vx0) ? 1.f : 0.f;  w11 *= (vy1 && vx1) ? 1.f : 0.f;
    int cy0 = min(max(y0, 0), H_ - 1),     cy1 = min(max(y0 + 1, 0), H_ - 1);
    int cx0 = min(max(x0, 0), W_ - 1),     cx1 = min(max(x0 + 1, 0), W_ - 1);
#pragma unroll
    for (int c = 0; c < 8; ++c) {
      int cin = g * 8 + c;
      const float* xp = x + (size_t)(b * 128 + cin) * (H_ * W_);
      float v = w00 * xp[cy0 * W_ + cx0] + w01 * xp[cy0 * W_ + cx1] +
                w10 * xp[cy1 * W_ + cx0] + w11 * xp[cy1 * W_ + cx1];
      val[p * 1152 + cin * 9 + k] = v * mv;
    }
  }
  __syncthreads();

  // phase 2: 64 couts x 16 pixels; thread = (o, pq), 4 pixels per thread
  int o  = tid & 63;
  int pq = tid >> 6;
  float a0 = 0.f, a1 = 0.f, a2 = 0.f, a3 = 0.f;
  const float* wr = wdcn + (size_t)o * 1152;
  const float* v0 = &val[(pq * 4 + 0) * 1152];
  const float* v1 = &val[(pq * 4 + 1) * 1152];
  const float* v2 = &val[(pq * 4 + 2) * 1152];
  const float* v3 = &val[(pq * 4 + 3) * 1152];
  for (int kk = 0; kk < 1152; kk += 4) {
    v4f wv = *(const v4f*)(wr + kk);
#pragma unroll
    for (int j2 = 0; j2 < 4; ++j2) {
      float wj = wv[j2];
      a0 += v0[kk + j2] * wj;
      a1 += v1[kk + j2] * wj;
      a2 += v2[kk + j2] * wj;
      a3 += v3[kk + j2] * wj;
    }
  }
  float bia = bdcn[o];
  size_t ob = ((size_t)(b * 64 + o) * H_ + h) * W_ + w0 + pq * 4;
  v4f res = {a0 + bia, a1 + bia, a2 + bia, a3 + bia};
  *(v4f*)&out[ob] = res;                          // 16B aligned contiguous store
}

// ---------------- launch ----------------
extern "C" void kernel_launch(void* const* d_in, const int* in_sizes, int n_in,
                              void* d_out, int out_size, void* d_ws, size_t ws_size,
                              hipStream_t stream) {
  const float* x    = (const float*)d_in[0];
  const float* ef   = (const float*)d_in[1];
  const float* f1   = (const float*)d_in[2];
  const float* f2   = (const float*)d_in[3];
  const float* w1   = (const float*)d_in[4];
  const float* b1   = (const float*)d_in[5];
  const float* w2   = (const float*)d_in[6];
  const float* b2   = (const float*)d_in[7];
  const float* w3   = (const float*)d_in[8];
  const float* b3   = (const float*)d_in[9];
  const float* w4   = (const float*)d_in[10];
  const float* b4   = (const float*)d_in[11];
  const float* dcnw = (const float*)d_in[12];
  const float* dcnb = (const float*)d_in[13];

  char* ws = (char*)d_ws;
  size_t cur = 0;
  auto alloc = [&](size_t bytes) {
    size_t r = cur;
    cur = (cur + bytes + 255) & ~(size_t)255;
    return r;
  };
  unsigned short* concat = (unsigned short*)(ws + alloc((size_t)B_ * H_ * W_ * C1PAD * 2));
  unsigned short* act1   = (unsigned short*)(ws + alloc((size_t)B_ * H_ * W_ * 64 * 2));
  unsigned short* act2   = (unsigned short*)(ws + alloc((size_t)B_ * H_ * W_ * 64 * 2));
  unsigned short* act3   = (unsigned short*)(ws + alloc((size_t)B_ * H_ * W_ * 64 * 2));
  float*          offsb  = (float*)(ws + alloc((size_t)B_ * H_ * W_ * 288 * 4));
  float*          maskb  = (float*)(ws + alloc((size_t)B_ * H_ * W_ * 144 * 4));
  unsigned short* wp1    = (unsigned short*)(ws + alloc((size_t)4 * 7 * 9 * 512 * 2));
  unsigned short* wp2    = (unsigned short*)(ws + alloc((size_t)4 * 2 * 9 * 512 * 2));
  unsigned short* wp3    = (unsigned short*)(ws + alloc((size_t)4 * 2 * 9 * 512 * 2));
  unsigned short* wp4    = (unsigned short*)(ws + alloc((size_t)MT4PAD * 2 * 9 * 512 * 2));

  {
    int total = B_ * H_ * W_ * C1PAD;
    pack_concat_kernel<<<(total + 255) / 256, 256, 0, stream>>>(ef, f1, f2, concat);
  }
  repack_w_kernel<<<(4 * 7 * 9 * 512 + 255) / 256, 256, 0, stream>>>(w1, wp1, 64, C1RAW, 7, 4);
  repack_w_kernel<<<(4 * 2 * 9 * 512 + 255) / 256, 256, 0, stream>>>(w2, wp2, 64, 64, 2, 4);
  repack_w_kernel<<<(4 * 2 * 9 * 512 + 255) / 256, 256, 0, stream>>>(w3, wp3, 64, 64, 2, 4);
  repack_w_kernel<<<(MT4PAD * 2 * 9 * 512 + 255) / 256, 256, 0, stream>>>(w4, wp4, COUT4, 64, 2, MT4PAD);

  dim3 grid1(B_ * H_ * (W_ / 32), 1);   // 1024 x 1
  conv_wmma_kernel<0><<<grid1, 256, 0, stream>>>(concat, wp1, b1, 7, 4, act1,
                                                 nullptr, nullptr, nullptr, nullptr);
  conv_wmma_kernel<0><<<grid1, 256, 0, stream>>>(act1, wp2, b2, 2, 4, act2,
                                                 nullptr, nullptr, nullptr, nullptr);
  conv_wmma_kernel<0><<<grid1, 256, 0, stream>>>(act2, wp3, b3, 2, 4, act3,
                                                 nullptr, nullptr, nullptr, nullptr);
  dim3 grid4(B_ * H_ * (W_ / 32), MT4PAD / 4);  // 1024 x 7
  conv_wmma_kernel<1><<<grid4, 256, 0, stream>>>(act3, wp4, b4, 2, MT4, nullptr,
                                                 offsb, maskb, f1, f2);

  dcn_kernel<<<B_ * H_ * (W_ / 16), 256, 0, stream>>>(x, offsb, maskb, dcnw, dcnb,
                                                      (float*)d_out);
}
